// LSTM_11020886081930
// MI455X (gfx1250) — matrix operand order repaired and checked by reading, not compiled, not run on hardware
//
#include <hip/hip_runtime.h>
#include <hip/hip_bf16.h>
#include <math.h>

// ---------------------------------------------------------------------------
// QLSTM (2 layers, B=64, S=2048, IN=H=256) for gfx1250 / MI455X.
// bf16 WMMA (v_wmma_f32_16x16x32_bf16) with fp32 accumulation + fp32 cell state.
// K0: convert weights to bf16 workspace copies; pre-sum biases in f32.
// K1: time-parallel GEMM xg0 = x @ W_ih0^T + (b_ih0+b_hh0)  (M=131072,N=768,K=256)
// K2: fused 2-layer sequential scan, 4 WGs (16 batch rows each), state in LDS.
// ---------------------------------------------------------------------------

typedef __attribute__((ext_vector_type(16))) __bf16 v16bf;
typedef __attribute__((ext_vector_type(8)))  __bf16 v8bf;
typedef __attribute__((ext_vector_type(8)))  float  v8f;

#define BATCH 64
#define SEQ   2048
#define HID   256
#define G3    768          // 3*H
#define AS    272          // LDS bf16 row stride (pad: 544B rows -> banks rotate)
#define HS    272          // LDS bf16 stride for h state
#define GS    776          // LDS f32  stride for gates (768 + 8 pad dwords)

__device__ __forceinline__ v16bf cat8(v8bf lo, v8bf hi) {
  return __builtin_shufflevector(lo, hi, 0,1,2,3,4,5,6,7,8,9,10,11,12,13,14,15);
}
__device__ __forceinline__ v8f wmma_bf16(v16bf a, v16bf b, v8f c) {
  return __builtin_amdgcn_wmma_f32_16x16x32_bf16(false, a, false, b, (short)0, c,
                                                 false, false);
}
__device__ __forceinline__ float sigmoid_f(float x) {
  return 1.0f / (1.0f + __expf(-x));
}

// ---------------------------------------------------------------------------
// K0: weights f32 -> bf16 ; bias0 = b_ih0+b_hh0 ; bias1 = b_ih1+b_hh1
// grid = 768 blocks (one gate-row each), 256 threads (one col each)
// ---------------------------------------------------------------------------
__global__ void k_convert(const float* __restrict__ Wih0,
                          const float* __restrict__ Whh0,
                          const float* __restrict__ Wih1,
                          const float* __restrict__ Whh1,
                          const float* __restrict__ bih0, const float* __restrict__ bhh0,
                          const float* __restrict__ bih1, const float* __restrict__ bhh1,
                          __bf16* __restrict__ wih0b, __bf16* __restrict__ whh0b,
                          __bf16* __restrict__ wih1b, __bf16* __restrict__ whh1b,
                          float* __restrict__ bias0, float* __restrict__ bias1) {
  const int row = blockIdx.x;
  const int t   = threadIdx.x;
  const int idx = row * HID + t;
  wih0b[idx] = (__bf16)Wih0[idx];
  whh0b[idx] = (__bf16)Whh0[idx];
  wih1b[idx] = (__bf16)Wih1[idx];
  whh1b[idx] = (__bf16)Whh1[idx];
  if (t == 0) {
    bias0[row] = bih0[row] + bhh0[row];
    bias1[row] = bih1[row] + bhh1[row];
  }
}

// ---------------------------------------------------------------------------
// K1: xg0[B*S][768] (bf16) = x[B*S][256] @ Wih0^T + bias0
// grid = 1024 blocks (128 M-rows each), 256 threads = 8 waves.
// Wave w owns M-strip [w*16, w*16+16), iterates all 48 N-tiles, K=256 in 8 chunks.
// A staged once to LDS as bf16; B read 32B-contiguous from bf16 weight copy.
// ---------------------------------------------------------------------------
__global__ void k_xgemm(const float* __restrict__ x,
                        const __bf16* __restrict__ Wb,
                        const float* __restrict__ bias0,
                        __bf16* __restrict__ xg) {
  __shared__ __attribute__((aligned(32))) __bf16 Alds[128 * AS];

  const int tid   = threadIdx.x;
  const int mbase = blockIdx.x * 128;

  // stage A (f32 -> bf16): 2 threads per row, 128 cols each
  {
    const int row = tid >> 1;
    const int c0  = (tid & 1) * 128;
    const float* src = x + (size_t)(mbase + row) * HID + c0;
    __bf16*      dst = Alds + row * AS + c0;
    for (int i = 0; i < 128; i += 4) {
      float4 v = *(const float4*)(src + i);
      dst[i + 0] = (__bf16)v.x;  dst[i + 1] = (__bf16)v.y;
      dst[i + 2] = (__bf16)v.z;  dst[i + 3] = (__bf16)v.w;
    }
  }
  __syncthreads();

  const int wave = tid >> 5, lane = tid & 31;
  const int r    = lane & 15;          // A row within strip / D column index
  const int half = lane >> 4;
  const int koff = half * 16;          // A: K sub-offset within 32-chunk
  const int kb   = half * 16;          // B: K base within 32-chunk
  const __bf16* arow = Alds + (wave * 16 + r) * AS + koff;

  for (int nt = 0; nt < 48; ++nt) {
    const int n0  = nt * 16;
    const int col = n0 + r;                                // this lane's N (B column)
    const __bf16* bp = Wb + (size_t)col * HID + kb;
    v8f acc = {0.f, 0.f, 0.f, 0.f, 0.f, 0.f, 0.f, 0.f};
    #pragma unroll
    for (int kc = 0; kc < 8; ++kc) {
      v8bf  lo = *(const v8bf*)(arow + kc * 32);
      v8bf  hi = *(const v8bf*)(arow + kc * 32 + 8);
      v16bf b  = *(const v16bf*)(bp + kc * 32);
      acc = wmma_bf16(cat8(lo, hi), b, acc);
    }
    const float bn = bias0[col];
    #pragma unroll
    for (int rr = 0; rr < 8; ++rr) {
      const size_t grow = (size_t)(mbase + wave * 16 + rr + 8 * half);
      xg[grow * G3 + col] = (__bf16)(acc[rr] + bn);
    }
  }
}

// ---------------------------------------------------------------------------
// K2: fused 2-layer scan. grid = 4 WGs (16 batch rows each), 256 thr = 8 waves.
// Per step: gates0 = xg0_t + h0@Whh0^T      -> pointwise -> h0,c0
//           gates1 = b1 + h0@Wih1^T + h1@Whh1^T -> pointwise -> h1,c1 (y out)
// h kept bf16 in LDS (WMMA A operand), c kept fp32 in LDS.
// ---------------------------------------------------------------------------
__global__ void k_scan(const __bf16* __restrict__ xg,
                       const __bf16* __restrict__ Whh0b,
                       const __bf16* __restrict__ Wih1b,
                       const __bf16* __restrict__ Whh1b,
                       const float* __restrict__ bias1g,
                       float* __restrict__ out) {
  __shared__ __attribute__((aligned(32))) __bf16 h0s[16 * HS];
  __shared__ __attribute__((aligned(32))) __bf16 h1s[16 * HS];
  __shared__ float c0s[16 * HID];
  __shared__ float c1s[16 * HID];
  __shared__ float gates[16 * GS];
  __shared__ float bias1[G3];

  const int tid = threadIdx.x;
  const int b0  = blockIdx.x * 16;

  for (int i = tid; i < 16 * HS; i += 256) { h0s[i] = (__bf16)0.f; h1s[i] = (__bf16)0.f; }
  for (int i = tid; i < 16 * HID; i += 256) { c0s[i] = 0.f; c1s[i] = 0.f; }
  for (int i = tid; i < G3; i += 256) bias1[i] = bias1g[i];
  __syncthreads();

  const int wave = tid >> 5, lane = tid & 31;
  const int r    = lane & 15;
  const int half = lane >> 4;
  const int koff = half * 16;
  const int kb   = half * 16;

  // pointwise mapping: thread -> (batch row pm, 16 hidden units from pj)
  const int pm = tid >> 4;
  const int pj = (tid & 15) * 16;

  const size_t outH = (size_t)BATCH * SEQ * HID;      // start of h_final
  const size_t outC = outH + (size_t)2 * BATCH * HID; // start of c_final

  for (int t = 0; t < SEQ; ++t) {
    // ---------------- layer 0 recurrent GEMM (K = 256) ----------------
    for (int i = 0; i < 6; ++i) {
      const int n0  = (wave * 6 + i) * 16;
      const int col = n0 + r;
      const __bf16* bp = Whh0b + (size_t)col * HID + kb;
      const __bf16* ap = h0s + r * HS + koff;
      v8f acc = {0.f, 0.f, 0.f, 0.f, 0.f, 0.f, 0.f, 0.f};
      #pragma unroll
      for (int kc = 0; kc < 8; ++kc) {
        v8bf  lo = *(const v8bf*)(ap + kc * 32);
        v8bf  hi = *(const v8bf*)(ap + kc * 32 + 8);
        v16bf b  = *(const v16bf*)(bp + kc * 32);
        acc = wmma_bf16(cat8(lo, hi), b, acc);
      }
      #pragma unroll
      for (int rr = 0; rr < 8; ++rr)
        gates[(rr + 8 * half) * GS + col] = acc[rr];
    }
    __syncthreads();

    // ---------------- pointwise layer 0 ----------------
    {
      const __bf16* xgrow = xg + ((size_t)(b0 + pm) * SEQ + t) * G3;
      if (t + 1 < SEQ) __builtin_prefetch(xgrow + G3, 0, 3);  // global_prefetch_b8
      #pragma unroll 4
      for (int j = pj; j < pj + 16; ++j) {
        const float ig = gates[pm * GS + j]             + (float)xgrow[j];
        const float gg = gates[pm * GS + HID + j]       + (float)xgrow[HID + j];
        const float og = gates[pm * GS + 2 * HID + j]   + (float)xgrow[2 * HID + j];
        const float cp = c0s[pm * HID + j];
        const float iv = sigmoid_f(ig);
        const float fv = sigmoid_f(cp);          // QLSTM quirk: f = sigmoid(c_prev)
        const float gv = tanhf(gg);
        const float ov = sigmoid_f(og);
        const float cn = fv * cp + 2.0f * iv * gv;
        const float hn = ov * tanhf(cn);
        c0s[pm * HID + j] = cn;
        h0s[pm * HS + j]  = (__bf16)hn;
        if (t == SEQ - 1) {
          out[outH + (size_t)(b0 + pm) * HID + j] = hn;   // h_final layer 0
          out[outC + (size_t)(b0 + pm) * HID + j] = cn;   // c_final layer 0
        }
      }
    }
    __syncthreads();

    // ------- layer 1 GEMM (K = 512: input proj on h0_new + recurrent) -------
    for (int i = 0; i < 6; ++i) {
      const int n0  = (wave * 6 + i) * 16;
      const int col = n0 + r;
      const __bf16* bpA = Wih1b + (size_t)col * HID + kb;
      const __bf16* bpB = Whh1b + (size_t)col * HID + kb;
      const __bf16* ap0 = h0s + r * HS + koff;
      const __bf16* ap1 = h1s + r * HS + koff;
      v8f acc = {0.f, 0.f, 0.f, 0.f, 0.f, 0.f, 0.f, 0.f};
      #pragma unroll
      for (int kc = 0; kc < 8; ++kc) {
        v8bf  lo = *(const v8bf*)(ap0 + kc * 32);
        v8bf  hi = *(const v8bf*)(ap0 + kc * 32 + 8);
        v16bf b  = *(const v16bf*)(bpA + kc * 32);
        acc = wmma_bf16(cat8(lo, hi), b, acc);
      }
      #pragma unroll
      for (int kc = 0; kc < 8; ++kc) {
        v8bf  lo = *(const v8bf*)(ap1 + kc * 32);
        v8bf  hi = *(const v8bf*)(ap1 + kc * 32 + 8);
        v16bf b  = *(const v16bf*)(bpB + kc * 32);
        acc = wmma_bf16(cat8(lo, hi), b, acc);
      }
      #pragma unroll
      for (int rr = 0; rr < 8; ++rr)
        gates[(rr + 8 * half) * GS + col] = acc[rr];
    }
    __syncthreads();

    // ---------------- pointwise layer 1 (+ y output) ----------------
    {
      float* yrow = out + ((size_t)(b0 + pm) * SEQ + t) * HID;
      #pragma unroll 4
      for (int j = pj; j < pj + 16; ++j) {
        const float ig = gates[pm * GS + j]           + bias1[j];
        const float gg = gates[pm * GS + HID + j]     + bias1[HID + j];
        const float og = gates[pm * GS + 2 * HID + j] + bias1[2 * HID + j];
        const float cp = c1s[pm * HID + j];
        const float iv = sigmoid_f(ig);
        const float fv = sigmoid_f(cp);
        const float gv = tanhf(gg);
        const float ov = sigmoid_f(og);
        const float cn = fv * cp + 2.0f * iv * gv;
        const float hn = ov * tanhf(cn);
        c1s[pm * HID + j] = cn;
        h1s[pm * HS + j]  = (__bf16)hn;
        yrow[j] = hn;
        if (t == SEQ - 1) {
          out[outH + (size_t)BATCH * HID + (size_t)(b0 + pm) * HID + j] = hn;
          out[outC + (size_t)BATCH * HID + (size_t)(b0 + pm) * HID + j] = cn;
        }
      }
    }
    __syncthreads();
  }
}

// ---------------------------------------------------------------------------
// Workspace layout (bytes):
//   0        Wih0 bf16   (393216)
//   393216   Whh0 bf16
//   786432   Wih1 bf16
//   1179648  Whh1 bf16
//   1572864  bias0 f32   (3072)
//   1575936  bias1 f32   (3072)
//   2097152  xg0 bf16    (64*2048*768*2 = 192 MiB)
// ---------------------------------------------------------------------------
extern "C" void kernel_launch(void* const* d_in, const int* in_sizes, int n_in,
                              void* d_out, int out_size, void* d_ws, size_t ws_size,
                              hipStream_t stream) {
  const float* x    = (const float*)d_in[0];
  const float* Wih0 = (const float*)d_in[1];
  const float* Whh0 = (const float*)d_in[2];
  const float* bih0 = (const float*)d_in[3];
  const float* bhh0 = (const float*)d_in[4];
  const float* Wih1 = (const float*)d_in[5];
  const float* Whh1 = (const float*)d_in[6];
  const float* bih1 = (const float*)d_in[7];
  const float* bhh1 = (const float*)d_in[8];

  char* ws = (char*)d_ws;
  __bf16* wih0b = (__bf16*)(ws + 0);
  __bf16* whh0b = (__bf16*)(ws + 393216);
  __bf16* wih1b = (__bf16*)(ws + 786432);
  __bf16* whh1b = (__bf16*)(ws + 1179648);
  float*  bias0 = (float*)(ws + 1572864);
  float*  bias1 = (float*)(ws + 1575936);
  __bf16* xg    = (__bf16*)(ws + 2097152);

  float* out = (float*)d_out;

  k_convert<<<dim3(G3), dim3(256), 0, stream>>>(Wih0, Whh0, Wih1, Whh1,
                                                bih0, bhh0, bih1, bhh1,
                                                wih0b, whh0b, wih1b, whh1b,
                                                bias0, bias1);
  k_xgemm<<<dim3((BATCH * SEQ) / 128), dim3(256), 0, stream>>>(x, wih0b, bias0, xg);
  k_scan<<<dim3(BATCH / 16), dim3(256), 0, stream>>>(xg, whh0b, wih1b, whh1b,
                                                     bias1, out);
}
